// LAUCell_51427938402962
// MI455X (gfx1250) — compile-verified, matrix operand order checked by reading, not run on hardware
//
#include <hip/hip_runtime.h>
#include <hip/hip_bf16.h>
#include <math.h>

// ---------------------------------------------------------------------------
// Dimensions (fixed by the reference)
// ---------------------------------------------------------------------------
#define NINP  1024
#define NHID  1024
#define BATCH 16384
#define NCOL  9216              // [ig 3072 | Hx 1024 | xh 1024 | hg 3072 | hh 1024]
#define LN_EPS 1e-5f

typedef __attribute__((ext_vector_type(16))) __bf16 v16bf;
typedef __attribute__((ext_vector_type(8)))  __bf16 v8bf;
typedef __attribute__((ext_vector_type(8)))  float  v8f;

union V16U { v16bf v; v8bf h[2]; };

__device__ inline v8bf pack8(const float4 a, const float4 b) {
    v8bf r;
    r[0] = (__bf16)a.x; r[1] = (__bf16)a.y; r[2] = (__bf16)a.z; r[3] = (__bf16)a.w;
    r[4] = (__bf16)b.x; r[5] = (__bf16)b.y; r[6] = (__bf16)b.z; r[7] = (__bf16)b.w;
    return r;
}

// gfx1250 async DMA: global -> LDS, 16B per lane, tracked by ASYNCcnt.
__device__ inline void async_load_b128(unsigned lds_off, const void* gaddr) {
    asm volatile("global_load_async_to_lds_b128 %0, %1, off"
                 :: "v"(lds_off), "v"((unsigned long long)(uintptr_t)gaddr)
                 : "memory");
}
__device__ inline void wait_async0() {
    asm volatile("s_wait_asynccnt 0x0" ::: "memory");
}

// ---------------------------------------------------------------------------
// Kernel 0: f32 -> bf16 conversion (inp, hidden, and packed weights)
// ---------------------------------------------------------------------------
__global__ __launch_bounds__(256)
void cvt_f32_bf16(const float* __restrict__ src, __bf16* __restrict__ dst) {
    const size_t i = ((size_t)blockIdx.x * 256 + threadIdx.x) * 8;
    const float4* s = (const float4*)(src + i);
    *(v8bf*)(dst + i) = pack8(s[0], s[1]);
}

// ---------------------------------------------------------------------------
// Kernel 1: fused 5-matmul bf16 WMMA GEMM, async-DMA staged, double-buffered.
//   C[b, col] (f32), col in the concatenated N space. Wcat is the packed
//   weight matrix in that same column order, so B has a single base pointer.
//   Block tile: 128(M) x 128(N) x 32(K). 256 threads = 8 wave32 (2x4).
// ---------------------------------------------------------------------------
#define LDT 40   // padded LDS row stride in bf16 (80B: conflict-free, 16B aligned)

__global__ __launch_bounds__(256)
void lau_wmma_gemm(const __bf16* __restrict__ xb,
                   const __bf16* __restrict__ hb,
                   const __bf16* __restrict__ Wcat,
                   float* __restrict__ C) {
    __shared__ __attribute__((aligned(16))) __bf16 As[2][128 * LDT];
    __shared__ __attribute__((aligned(16))) __bf16 Bs[2][128 * LDT];

    const int colBase = blockIdx.x * 128;
    const int rowBase = blockIdx.y * 128;

    // inp feeds cols [0,5120) (ig|Hx|xh); hidden feeds [5120,9216) (hg|hh)
    const __bf16* A = (colBase < 5120) ? xb : hb;

    const int t    = threadIdx.x;
    const int lane = t & 31;
    const int wid  = t >> 5;          // 0..7
    const int wm   = wid >> 2;        // 0..1 : 64-row strip
    const int wn   = wid & 3;         // 0..3 : 32-col strip

    // DMA staging: thread owns 16 contiguous bf16 (two b128) of one tile row
    const int ra = t >> 1;            // 0..127 local row
    const int ck = (t & 1) * 16;      // element offset 0 or 16 in 32-wide K slab

    const __bf16* aSrc = A    + (size_t)(rowBase + ra) * NINP + ck;
    const __bf16* bSrc = Wcat + (size_t)(colBase + ra) * NINP + ck;

    unsigned aOff[2], bOff[2];
    aOff[0] = (unsigned)(uintptr_t)&As[0][ra * LDT + ck];
    aOff[1] = (unsigned)(uintptr_t)&As[1][ra * LDT + ck];
    bOff[0] = (unsigned)(uintptr_t)&Bs[0][ra * LDT + ck];
    bOff[1] = (unsigned)(uintptr_t)&Bs[1][ra * LDT + ck];

    v8f acc[4][2] = {};               // 4 (M sub-tiles) x 2 (N sub-tiles)

    const int nK = NINP / 32;
    // prologue: DMA tile 0 into buffer 0
    async_load_b128(aOff[0],      aSrc);
    async_load_b128(aOff[0] + 16, aSrc + 8);
    async_load_b128(bOff[0],      bSrc);
    async_load_b128(bOff[0] + 16, bSrc + 8);

    #pragma unroll 1                 // keep one body: fixed acc registers,
    for (int kb = 0; kb < nK; ++kb) {//  no cross-iteration copies/hazard NOPs
        const int p = kb & 1;
        const int q = p ^ 1;
        // branchless next-K: last iteration re-issues current tile into the
        // dead buffer (never read) instead of branching around the DMA.
        const int kn = (kb + 1 < nK) ? (kb + 1) * 32 : kb * 32;

        wait_async0();                // my buffer-p DMA complete
        __syncthreads();              // whole tile in LDS; buffer q reads done

        async_load_b128(aOff[q],      aSrc + kn);
        async_load_b128(aOff[q] + 16, aSrc + kn + 8);
        async_load_b128(bOff[q],      bSrc + kn);
        async_load_b128(bOff[q] + 16, bSrc + kn + 8);

        // ---- fragment loads (ISA 16-bit A/B layout) -----------------------
        const int k0 = (lane & 16) ? 8 : 0;   // lanes 16..31: K 8..15 / 24..31
        V16U a[4], b[2];
        #pragma unroll
        for (int tm = 0; tm < 4; ++tm) {
            const int r = (wm * 64 + tm * 16 + (lane & 15)) * LDT;
            a[tm].h[0] = *(const v8bf*)&As[p][r + k0];
            a[tm].h[1] = *(const v8bf*)&As[p][r + k0 + 16];
        }
        #pragma unroll
        for (int tn = 0; tn < 2; ++tn) {
            const int r = (wn * 32 + tn * 16 + (lane & 15)) * LDT;
            b[tn].h[0] = *(const v8bf*)&Bs[p][r + k0];
            b[tn].h[1] = *(const v8bf*)&Bs[p][r + k0 + 16];
        }

        // ---- 8x v_wmma_f32_16x16x32_bf16 ----------------------------------
        #pragma unroll
        for (int tn = 0; tn < 2; ++tn)
            #pragma unroll
            for (int tm = 0; tm < 4; ++tm)
                acc[tm][tn] = __builtin_amdgcn_wmma_f32_16x16x32_bf16(
                    false, a[tm].v, false, b[tn].v,
                    (short)0, acc[tm][tn], false, false);
    }

    // ---- write C tile: VGPR i holds M=i (lanes 0-15) / M=8+i (lanes 16-31)
    const int mOff = (lane & 16) ? 8 : 0;
    const int nOff = lane & 15;
    #pragma unroll
    for (int tm = 0; tm < 4; ++tm) {
        #pragma unroll
        for (int tn = 0; tn < 2; ++tn) {
            const size_t col = (size_t)(colBase + wn * 32 + tn * 16 + nOff);
            #pragma unroll
            for (int i = 0; i < 8; ++i) {
                const size_t row = (size_t)(rowBase + wm * 64 + tm * 16 + mOff + i);
                C[row * NCOL + col] = acc[tm][tn][i];
            }
        }
    }
}

// ---------------------------------------------------------------------------
// Kernel 2: per-row layernorms + gate fusion. One 256-thread block per row.
// ---------------------------------------------------------------------------
__device__ inline float sigmoidf_(float x) { return 1.0f / (1.0f + __expf(-x)); }

__global__ __launch_bounds__(256)
void lau_row_fuse(const float* __restrict__ ws,
                  const float* __restrict__ hidden,
                  const float* __restrict__ b_ig, const float* __restrict__ b_hg,
                  const float* __restrict__ b_ii, const float* __restrict__ b_ih,
                  const float* __restrict__ b_hh,
                  const float* __restrict__ g_ig, const float* __restrict__ be_ig,
                  const float* __restrict__ g_hg, const float* __restrict__ be_hg,
                  const float* __restrict__ g_ii, const float* __restrict__ be_ii,
                  const float* __restrict__ g_ih, const float* __restrict__ be_ih,
                  const float* __restrict__ g_hh, const float* __restrict__ be_hh,
                  float* __restrict__ out) {
    const int row  = blockIdx.x;
    const int t    = threadIdx.x;
    const int lane = t & 31;
    const int wid  = t >> 5;
    const float* wsr = ws + (size_t)row * NCOL;

    float ig[3][4], hg[3][4], hx[4], xh[4], hh[4];
    float p[10];
    #pragma unroll
    for (int q = 0; q < 10; ++q) p[q] = 0.0f;

    #pragma unroll
    for (int ii = 0; ii < 4; ++ii) {
        const int c = t + ii * 256;           // 0..1023
        #pragma unroll
        for (int m = 0; m < 3; ++m) {
            const int col = m * 1024 + c;
            float vi = wsr[col]        + b_ig[col];
            float vh = wsr[5120 + col] + b_hg[col];
            ig[m][ii] = vi; hg[m][ii] = vh;
            p[0] += vi; p[1] += vi * vi;
            p[2] += vh; p[3] += vh * vh;
        }
        float a = wsr[3072 + c] + b_ii[c];
        float b = wsr[4096 + c] + b_ih[c];
        float d = wsr[8192 + c] + b_hh[c];
        hx[ii] = a; xh[ii] = b; hh[ii] = d;
        p[4] += a; p[5] += a * a;
        p[6] += b; p[7] += b * b;
        p[8] += d; p[9] += d * d;
    }

    __shared__ float red[8][10];
    #pragma unroll
    for (int q = 0; q < 10; ++q) {
        float v = p[q];
        #pragma unroll
        for (int off = 16; off > 0; off >>= 1) v += __shfl_down(v, off, 32);
        if (lane == 0) red[wid][q] = v;
    }
    __syncthreads();
    float tot[10];
    #pragma unroll
    for (int q = 0; q < 10; ++q) {
        float s = 0.0f;
        #pragma unroll
        for (int w = 0; w < 8; ++w) s += red[w][q];
        tot[q] = s;
    }

    const float i3 = 1.0f / 3072.0f, i1 = 1.0f / 1024.0f;
    const float mu_ig = tot[0] * i3, rs_ig = rsqrtf(tot[1] * i3 - mu_ig * mu_ig + LN_EPS);
    const float mu_hg = tot[2] * i3, rs_hg = rsqrtf(tot[3] * i3 - mu_hg * mu_hg + LN_EPS);
    const float mu_hx = tot[4] * i1, rs_hx = rsqrtf(tot[5] * i1 - mu_hx * mu_hx + LN_EPS);
    const float mu_xh = tot[6] * i1, rs_xh = rsqrtf(tot[7] * i1 - mu_xh * mu_xh + LN_EPS);
    const float mu_hh = tot[8] * i1, rs_hh = rsqrtf(tot[9] * i1 - mu_hh * mu_hh + LN_EPS);

    #pragma unroll
    for (int ii = 0; ii < 4; ++ii) {
        const int c = t + ii * 256;
        float gate[3];
        #pragma unroll
        for (int m = 0; m < 3; ++m) {
            const int col = m * 1024 + c;
            float ni = (ig[m][ii] - mu_ig) * rs_ig * g_ig[col] + be_ig[col];
            float nh = (hg[m][ii] - mu_hg) * rs_hg * g_hg[col] + be_hg[col];
            gate[m] = sigmoidf_(ni + nh);
        }
        const float r = gate[0], z = gate[1], g = gate[2];
        const float Hx  = (hx[ii] - mu_hx) * rs_hx * g_ii[c] + be_ii[c];
        const float xhn = (xh[ii] - mu_xh) * rs_xh * g_ih[c] + be_ih[c];
        const float hhn = (hh[ii] - mu_hh) * rs_hh * g_hh[c] + be_hh[c];
        const float hm  = tanhf((1.0f - r) * xhn + r * hhn);
        const float h   = hidden[(size_t)row * NHID + c];
        out[(size_t)row * NHID + c] = ((1.0f - z) * h + z * hm) * (1.0f - g) + g * Hx;
    }
}

// ---------------------------------------------------------------------------
// Host launcher
// ---------------------------------------------------------------------------
extern "C" void kernel_launch(void* const* d_in, const int* in_sizes, int n_in,
                              void* d_out, int out_size, void* d_ws, size_t ws_size,
                              hipStream_t stream) {
    (void)in_sizes; (void)n_in; (void)out_size; (void)ws_size;
    const float* inp   = (const float*)d_in[0];
    const float* hid   = (const float*)d_in[1];
    const float* Wig   = (const float*)d_in[2];
    const float* b_ig  = (const float*)d_in[3];
    const float* Whg   = (const float*)d_in[4];
    const float* b_hg  = (const float*)d_in[5];
    const float* Wii   = (const float*)d_in[6];
    const float* b_ii  = (const float*)d_in[7];
    const float* Wih   = (const float*)d_in[8];
    const float* b_ih  = (const float*)d_in[9];
    const float* Whh   = (const float*)d_in[10];
    const float* b_hh  = (const float*)d_in[11];
    const float* g_ig  = (const float*)d_in[12];
    const float* be_ig = (const float*)d_in[13];
    const float* g_hg  = (const float*)d_in[14];
    const float* be_hg = (const float*)d_in[15];
    const float* g_ii  = (const float*)d_in[16];
    const float* be_ii = (const float*)d_in[17];
    const float* g_ih  = (const float*)d_in[18];
    const float* be_ih = (const float*)d_in[19];
    const float* g_hh  = (const float*)d_in[20];
    const float* be_hh = (const float*)d_in[21];

    // Workspace layout (bytes):
    //   C (f32)    : BATCH*NCOL*4            = 603,979,776
    //   xb (bf16)  : BATCH*NINP*2            =  33,554,432
    //   hb (bf16)  : BATCH*NHID*2            =  33,554,432
    //   Wcat (bf16): NCOL*NINP*2             =  18,874,368   (~658 MB total)
    float*  C  = (float*)d_ws;
    __bf16* xb = (__bf16*)((char*)d_ws + (size_t)BATCH * NCOL * 4);
    __bf16* hb = xb + (size_t)BATCH * NINP;
    __bf16* Wc = hb + (size_t)BATCH * NHID;
    float*  out = (float*)d_out;

    // phase 0: bf16 conversion + weight packing into concatenated order
    const int TPB = 256, EPT = 8;
    cvt_f32_bf16<<<(BATCH * NINP) / (TPB * EPT), TPB, 0, stream>>>(inp, xb);
    cvt_f32_bf16<<<(BATCH * NHID) / (TPB * EPT), TPB, 0, stream>>>(hid, hb);
    cvt_f32_bf16<<<(3072 * NINP) / (TPB * EPT), TPB, 0, stream>>>(Wig, Wc);
    cvt_f32_bf16<<<(1024 * NINP) / (TPB * EPT), TPB, 0, stream>>>(Wii, Wc + (size_t)3072 * NINP);
    cvt_f32_bf16<<<(1024 * NINP) / (TPB * EPT), TPB, 0, stream>>>(Wih, Wc + (size_t)4096 * NINP);
    cvt_f32_bf16<<<(3072 * NINP) / (TPB * EPT), TPB, 0, stream>>>(Whg, Wc + (size_t)5120 * NINP);
    cvt_f32_bf16<<<(1024 * NINP) / (TPB * EPT), TPB, 0, stream>>>(Whh, Wc + (size_t)8192 * NINP);

    // phase 1: WMMA GEMM (async-DMA staged, double-buffered)
    dim3 g1(NCOL / 128, BATCH / 128);   // 72 x 128 tiles
    lau_wmma_gemm<<<g1, dim3(256), 0, stream>>>(xb, hb, Wc, C);

    // phase 2: row-wise LN + gate fusion
    lau_row_fuse<<<dim3(BATCH), dim3(256), 0, stream>>>(
        C, hid, b_ig, b_hg, b_ii, b_ih, b_hh,
        g_ig, be_ig, g_hg, be_hg, g_ii, be_ii, g_ih, be_ih, g_hh, be_hh, out);
}
